// GRUClassifier_78872779424239
// MI455X (gfx1250) — compile-verified, matrix operand order checked by reading, not run on hardware
//
#include <hip/hip_runtime.h>

// ---------------------------------------------------------------------------
// Types for CDNA5 WMMA (wave32): bf16 16x16x32, f32 accumulate.
// ---------------------------------------------------------------------------
typedef __attribute__((ext_vector_type(16))) __bf16 bf16x16;
typedef __attribute__((ext_vector_type(8)))  float  f32x8;

union FragBF {
    uint4   u[2];   // two 16-byte chunks (8 bf16 each)
    bf16x16 v;
};

union F8 {
    float4 f4[2];
    float  f[8];
};

__device__ __forceinline__ f32x8 wmma_bf16(bf16x16 a, bf16x16 b, f32x8 c) {
    // (neg_a, A, neg_b, B, c_mod, C, reuse_a, reuse_b)
    return __builtin_amdgcn_wmma_f32_16x16x32_bf16(false, a, false, b, (short)0, c,
                                                   false, false);
}

__device__ __forceinline__ f32x8 zero8() {
    f32x8 z;
#pragma unroll
    for (int i = 0; i < 8; ++i) z[i] = 0.0f;
    return z;
}

__device__ __forceinline__ unsigned short f32_to_bf16(float f) {
    unsigned u = __float_as_uint(f);
    unsigned r = u + 0x7FFFu + ((u >> 16) & 1u);   // round-to-nearest-even
    return (unsigned short)(r >> 16);
}

__device__ __forceinline__ float sigmoidf_fast(float x) {
    return 1.0f / (1.0f + __expf(-x));
}

// Problem dims
#define TT 512
#define BB 64
#define HH 512
#define N3 1536
#define EE 256
#define MTOT ((long)TT * BB)   // 32768 rows per direction

// LDS row padding: 512 + 8 bf16 (=16B) -> row stride 1040B, 4-bank rotation/row
#define HPAD 520
// LDS layout: weights [3*64][HPAD] bf16, then h [64][HPAD] bf16
#define LDS_W_ROWS 192
#define LDS_H_OFF  (LDS_W_ROWS * HPAD)
#define LDS_BYTES  ((LDS_W_ROWS + 64) * HPAD * 2)

// ---------------------------------------------------------------------------
// Tensor Data Mover (gfx1250): one descriptor DMAs a padded tile into LDS.
// D# packing per CDNA5 ISA. Guarded by __has_builtin.
// This toolchain's builtin is the 6-arg form:
//   (uint32x4 g0, int32x8 g1, int32x4 g2, int32x4 g3, int32x8 g4, i32 cpol)
// ---------------------------------------------------------------------------
#if __has_builtin(__builtin_amdgcn_tensor_load_to_lds) && \
    __has_builtin(__builtin_amdgcn_s_wait_tensorcnt)
#define USE_TDM 1
typedef __attribute__((ext_vector_type(4))) unsigned int u32x4;
typedef __attribute__((ext_vector_type(8))) int i32x8;
typedef __attribute__((ext_vector_type(4))) int i32x4;

// Load tile of bf16 rows (512 elems/row -> 1024B) from global into LDS with
// 16B padding appended per row (matches HPAD=520 layout).
//   tile_rows : rows per chunk (tile_dim1)
//   z_chunks  : number of row-chunks (tile_dim2; 0 = plain 2D)
//   z_stride  : global element stride between chunks (tensor_dim1_stride)
__device__ __forceinline__ void tdm_load_tile(unsigned lds_byte_addr,
                                              const unsigned short* gsrc,
                                              int tile_rows, int z_chunks,
                                              long z_stride) {
    unsigned long long ga = (unsigned long long)(size_t)gsrc;
    u32x4 g0;
    g0[0] = 1u;                                            // count=1, user mode
    g0[1] = lds_byte_addr;                                 // lds_addr [63:32]
    g0[2] = (unsigned)(ga & 0xFFFFFFFFull);                // global_addr lo
    g0[3] = (unsigned)((ga >> 32) & 0x01FFFFFFull)         // global_addr hi
            | (2u << 30);                                  // type = 2 (image)
    i32x8 g1;
    g1[0] = (1 << 16)      // data_size = 2 bytes
          | (1 << 20)      // pad_enable
          | (7 << 22)      // pad_interval = 256 DWORDs (one 1024B row)
          | (3 << 25);     // pad_amount = 4 DWORDs (16B)
    g1[1] = (512 & 0xFFFF) << 16;                 // tensor_dim0 lo16 (=512)
    g1[2] = ((N3 & 0xFFFF) << 16);                // dim0 hi=0 | tensor_dim1 lo16
    g1[3] = (512 << 16);                          // dim1 hi=0 | tile_dim0 = 512
    g1[4] = tile_rows | (z_chunks << 16);         // tile_dim1 | tile_dim2
    g1[5] = 512;                                  // tensor_dim0_stride lo32
    g1[6] = (int)((z_stride & 0xFFFFll) << 16);   // stride0 hi=0 | dim1_stride lo16
    g1[7] = (int)(z_stride >> 16);                // dim1_stride [47:16]
    i32x4 g2;
    g2[0] = 8;  g2[1] = 0;  g2[2] = 0;  g2[3] = 0;   // tensor_dim2 (bound only)
    i32x4 g3;
    g3[0] = 0;  g3[1] = 0;  g3[2] = 0;  g3[3] = 0;
    i32x8 g4;
#pragma unroll
    for (int i = 0; i < 8; ++i) g4[i] = 0;
    __builtin_amdgcn_tensor_load_to_lds(g0, g1, g2, g3, g4, 0);
}
#else
#define USE_TDM 0
#endif

// ---------------------------------------------------------------------------
// fp32 -> bf16 elementwise convert (weights)
// ---------------------------------------------------------------------------
__global__ void cvt_bf16_kernel(const float* __restrict__ src,
                                unsigned short* __restrict__ dst, int n) {
    int i = blockIdx.x * blockDim.x + threadIdx.x;
    int stride = gridDim.x * blockDim.x;
    for (; i < n; i += stride) dst[i] = f32_to_bf16(src[i]);
}

// ---------------------------------------------------------------------------
// Embedding gather -> x0 bf16 [T][B][E]  (row = t*64+b)
// ---------------------------------------------------------------------------
__global__ void embed_gather_kernel(const int* __restrict__ sent,
                                    const float* __restrict__ emb,
                                    unsigned short* __restrict__ x0) {
    int tid = blockIdx.x * blockDim.x + threadIdx.x;  // 32768*32 threads, 8 elems each
    int row = tid >> 5;
    if (row >= TT * BB) return;
    int e0 = (tid & 31) * 8;
    int t = row >> 6, b = row & 63;
    int widx = sent[b * TT + t];
    const float4* src = (const float4*)(emb + (long)widx * EE + e0);
    float4 lo = src[0], hi = src[1];
    uint4 packed;
    packed.x = (unsigned)f32_to_bf16(lo.x) | ((unsigned)f32_to_bf16(lo.y) << 16);
    packed.y = (unsigned)f32_to_bf16(lo.z) | ((unsigned)f32_to_bf16(lo.w) << 16);
    packed.z = (unsigned)f32_to_bf16(hi.x) | ((unsigned)f32_to_bf16(hi.y) << 16);
    packed.w = (unsigned)f32_to_bf16(hi.z) | ((unsigned)f32_to_bf16(hi.w) << 16);
    *(uint4*)(x0 + (long)row * EE + e0) = packed;
}

// ---------------------------------------------------------------------------
// Input projection GEMM: xg[dir][n][m] = sum_k A[m][k] * W[dir][n][k] + bias[dir][n]
// A: [M][K] bf16 ; W: [2][1536][K] bf16 ; out COLUMN-major: [2][1536][M] f32
// ---------------------------------------------------------------------------
__global__ __launch_bounds__(256) void gemm_inproj_kernel(
    const unsigned short* __restrict__ A, const unsigned short* __restrict__ W,
    const float* __restrict__ bias, float* __restrict__ out, int M, int K) {
    const int dir = blockIdx.z;
    W += (long)dir * N3 * K;
    bias += (long)dir * N3;
    out += (long)dir * M * N3;

    const int mbase = blockIdx.x * 64;
    const int wave = threadIdx.x >> 5;
    const int lane = threadIdx.x & 31;
    const int nbase = blockIdx.y * 128 + wave * 16;
    const int lm = lane & 15;
    const int khalf = (lane >> 4) * 8;

    f32x8 acc[4];
#pragma unroll
    for (int mt = 0; mt < 4; ++mt) acc[mt] = zero8();

    const unsigned short* wrow = W + (long)(nbase + lm) * K + khalf;
    for (int k = 0; k < K; k += 32) {
        FragBF fb;
        fb.u[0] = *(const uint4*)(wrow + k);
        fb.u[1] = *(const uint4*)(wrow + k + 16);
#pragma unroll
        for (int mt = 0; mt < 4; ++mt) {
            const unsigned short* ap = A + (long)(mbase + mt * 16 + lm) * K + k + khalf;
            FragBF fa;
            fa.u[0] = *(const uint4*)(ap);
            fa.u[1] = *(const uint4*)(ap + 16);
            acc[mt] = wmma_bf16(fa.v, fb.v, acc[mt]);
        }
    }

    const int n = nbase + lm;
    const float bv = bias[n];
    float* ocol = out + (long)n * M;
#pragma unroll
    for (int mt = 0; mt < 4; ++mt) {
        const int m0 = mbase + mt * 16 + (lane >> 4) * 8;
        float4 f0 = make_float4(acc[mt][0] + bv, acc[mt][1] + bv, acc[mt][2] + bv,
                                acc[mt][3] + bv);
        float4 f1 = make_float4(acc[mt][4] + bv, acc[mt][5] + bv, acc[mt][6] + bv,
                                acc[mt][7] + bv);
        *(float4*)(ocol + m0) = f0;
        *(float4*)(ocol + m0 + 4) = f1;
    }
}

// ---------------------------------------------------------------------------
// Zero init: barrier + ping-pong hidden state (f32 + bf16)
// ---------------------------------------------------------------------------
__global__ void init_zero_kernel(uint4* __restrict__ p, int n16) {
    int i = blockIdx.x * blockDim.x + threadIdx.x;
    if (i < n16) { uint4 z; z.x = z.y = z.z = z.w = 0u; p[i] = z; }
}

// ---------------------------------------------------------------------------
// Device-wide barrier (per direction, 8 co-resident workgroups)
// ---------------------------------------------------------------------------
__device__ __forceinline__ void grid_sync(unsigned* bar, unsigned nb) {
    __threadfence();
    __syncthreads();
    if (threadIdx.x == 0) {
        volatile unsigned* gen = bar + 16;
        unsigned g = *gen;
        if (atomicAdd(bar, 1u) == nb - 1u) {
            bar[0] = 0u;
            __threadfence();
            *gen = g + 1u;
        } else {
            while (*gen == g) __builtin_amdgcn_s_sleep(1);
        }
    }
    __syncthreads();
}

// ---------------------------------------------------------------------------
// Persistent GRU recurrence (one layer, both directions concurrently).
// grid = (8, 2): blockIdx.y = direction, 8 blocks x 4 waves x 16 cols = 512 r-cols.
// w_hh slice (3x64x512 bf16) staged in LDS ONCE via TDM (3D tile); shared h
// (64x512 bf16) staged via TDM each step. Inner loop: ds_load_b128 -> v_wmma.
// ---------------------------------------------------------------------------
__global__ __launch_bounds__(128) void gru_recurrence_kernel(
    const float* __restrict__ xg, const unsigned short* __restrict__ whh,
    const float* __restrict__ bhh, float* __restrict__ h32,
    unsigned short* __restrict__ hbf, unsigned short* __restrict__ ybf,
    float* __restrict__ hT, unsigned* __restrict__ bar) {
    extern __shared__ unsigned short smem[];   // [192][HPAD] weights + [64][HPAD] h

    const int dir = blockIdx.y;
    const int wave = threadIdx.x >> 5;
    const int lane = threadIdx.x & 31;
    const int lm = lane & 15;
    const int khalf = (lane >> 4) * 8;
    const int jblock = blockIdx.x * 64;
    const int lj = wave * 16 + lm;     // local column within block (0..63)
    const int j = jblock + lj;         // global r-column (0..511)
    unsigned* mybar = bar + dir * 32;  // per-direction barrier (separate lines)

    // ---- stage w_hh slice into LDS (once) --------------------------------
#if USE_TDM
    if (threadIdx.x < 32) {
        unsigned lds_w = (unsigned)(size_t)(const void*)smem;
        const unsigned short* wsrc = whh + ((long)dir * N3 + jblock) * HH;
        // 3 chunks of 64 rows; chunk g starts g*HH rows later = g*HH*HH elems
        tdm_load_tile(lds_w, wsrc, 64, 3, (long)HH * HH);
        __builtin_amdgcn_s_wait_tensorcnt(0);
    }
#else
    {
        uint4* wl = (uint4*)smem;
        const int rowu4 = HPAD / 8;  // uint4 per padded row = 65
        for (int i = threadIdx.x; i < LDS_W_ROWS * 64; i += blockDim.x) {
            int row = i >> 6;
            int c8 = i & 63;
            int g = row >> 6;
            int r = row & 63;
            const unsigned short* src =
                whh + ((long)(dir * N3 + g * HH + jblock + r)) * HH + c8 * 8;
            wl[row * rowu4 + c8] = *(const uint4*)src;
        }
    }
#endif
    __syncthreads();

    const float bh0 = bhh[dir * N3 + j];
    const float bh1 = bhh[dir * N3 + HH + j];
    const float bh2 = bhh[dir * N3 + 2 * HH + j];

    const unsigned short* wl0 = smem + (0 * 64 + lj) * HPAD + khalf;
    const unsigned short* wl1 = smem + (1 * 64 + lj) * HPAD + khalf;
    const unsigned short* wl2 = smem + (2 * 64 + lj) * HPAD + khalf;
    unsigned short* hl = smem + LDS_H_OFF;

    for (int t = 0; t < TT; ++t) {
        const int par = t & 1;
        const unsigned short* hin = hbf + ((long)(par * 2 + dir) * BB) * HH;
        // ---- stage h (bf16, row-major [64][512]) into LDS ----------------
#if USE_TDM
        if (threadIdx.x < 32) {
            unsigned lds_h = (unsigned)(size_t)(const void*)hl;
            tdm_load_tile(lds_h, hin, 64, 0, 0);
            __builtin_amdgcn_s_wait_tensorcnt(0);
        }
#else
        {
            uint4* hldst = (uint4*)hl;
            const int rowu4 = HPAD / 8;
            for (int i = threadIdx.x; i < 64 * 64; i += blockDim.x) {
                int m = i >> 6, c8 = i & 63;
                hldst[m * rowu4 + c8] = *(const uint4*)(hin + (long)m * HH + c8 * 8);
            }
        }
#endif
        __syncthreads();

        f32x8 acc[3][4];
#pragma unroll
        for (int g = 0; g < 3; ++g)
#pragma unroll
            for (int mt = 0; mt < 4; ++mt) acc[g][mt] = zero8();

        for (int k = 0; k < HH; k += 32) {
            FragBF fw0, fw1, fw2;
            fw0.u[0] = *(const uint4*)(wl0 + k);
            fw0.u[1] = *(const uint4*)(wl0 + k + 16);
            fw1.u[0] = *(const uint4*)(wl1 + k);
            fw1.u[1] = *(const uint4*)(wl1 + k + 16);
            fw2.u[0] = *(const uint4*)(wl2 + k);
            fw2.u[1] = *(const uint4*)(wl2 + k + 16);
#pragma unroll
            for (int mt = 0; mt < 4; ++mt) {
                const unsigned short* ap = hl + (mt * 16 + lm) * HPAD + k + khalf;
                FragBF fa;
                fa.u[0] = *(const uint4*)(ap);
                fa.u[1] = *(const uint4*)(ap + 16);
                acc[0][mt] = wmma_bf16(fa.v, fw0.v, acc[0][mt]);
                acc[1][mt] = wmma_bf16(fa.v, fw1.v, acc[1][mt]);
                acc[2][mt] = wmma_bf16(fa.v, fw2.v, acc[2][mt]);
            }
        }

        const int ta = dir ? (TT - 1 - t) : t;
        const float* xgc0 = xg + ((long)(dir * N3 + j)) * MTOT + (long)ta * BB;
        const float* xgc1 = xg + ((long)(dir * N3 + HH + j)) * MTOT + (long)ta * BB;
        const float* xgc2 = xg + ((long)(dir * N3 + 2 * HH + j)) * MTOT + (long)ta * BB;
        const float* hoc = h32 + ((long)(par * 2 + dir) * HH + j) * BB;
        float* hoc_out = h32 + ((long)((par ^ 1) * 2 + dir) * HH + j) * BB;
        unsigned short* hbf_out = hbf + ((long)((par ^ 1) * 2 + dir) * BB) * HH;

        // prefetch next step's gate inputs while stores/barrier drain
        const int tn = dir ? (TT - 2 - t) : (t + 1);
        if (tn >= 0 && tn < TT) {
            __builtin_prefetch(xgc0 + ((long)tn - ta) * BB, 0, 1);
            __builtin_prefetch(xgc1 + ((long)tn - ta) * BB, 0, 1);
            __builtin_prefetch(xgc2 + ((long)tn - ta) * BB, 0, 1);
        }

#pragma unroll
        for (int mt = 0; mt < 4; ++mt) {
            const int m0 = mt * 16 + (lane >> 4) * 8;
            F8 xr, xz, xn, ho, hv;
            xr.f4[0] = *(const float4*)(xgc0 + m0);
            xr.f4[1] = *(const float4*)(xgc0 + m0 + 4);
            xz.f4[0] = *(const float4*)(xgc1 + m0);
            xz.f4[1] = *(const float4*)(xgc1 + m0 + 4);
            xn.f4[0] = *(const float4*)(xgc2 + m0);
            xn.f4[1] = *(const float4*)(xgc2 + m0 + 4);
            ho.f4[0] = *(const float4*)(hoc + m0);
            ho.f4[1] = *(const float4*)(hoc + m0 + 4);
#pragma unroll
            for (int v = 0; v < 8; ++v) {
                float r = sigmoidf_fast(xr.f[v] + acc[0][mt][v] + bh0);
                float z = sigmoidf_fast(xz.f[v] + acc[1][mt][v] + bh1);
                float ng = tanhf(xn.f[v] + r * (acc[2][mt][v] + bh2));
                hv.f[v] = (1.0f - z) * ng + z * ho.f[v];
            }
            *(float4*)(hoc_out + m0) = hv.f4[0];
            *(float4*)(hoc_out + m0 + 4) = hv.f4[1];
#pragma unroll
            for (int v = 0; v < 8; ++v) {
                unsigned short hb = f32_to_bf16(hv.f[v]);
                hbf_out[(long)(m0 + v) * HH + j] = hb;
                ybf[((long)ta * BB + m0 + v) * 1024 + dir * HH + j] = hb;
            }
            if (t == TT - 1) {
#pragma unroll
                for (int v = 0; v < 8; ++v)
                    hT[((long)dir * BB + m0 + v) * HH + j] = hv.f[v];
            }
        }
        grid_sync(mybar, gridDim.x);
    }
}

// ---------------------------------------------------------------------------
// Final FC: out[b][c] = cat(hT_f, hT_b)[b] . fc_w[c] + fc_b[c]
// ---------------------------------------------------------------------------
__global__ void fc_kernel(const float* __restrict__ hT, const float* __restrict__ fcw,
                          const float* __restrict__ fcb, float* __restrict__ out) {
    int idx = blockIdx.x * blockDim.x + threadIdx.x;
    if (idx >= BB * 10) return;
    int b = idx / 10, c = idx % 10;
    float acc = fcb[c];
    const float* hf = hT + (long)b * HH;
    const float* hb = hT + (long)(BB + b) * HH;
    const float* wf = fcw + (long)c * 1024;
    for (int k = 0; k < HH; k += 4) {
        float4 h4 = *(const float4*)(hf + k);
        float4 w4 = *(const float4*)(wf + k);
        acc += h4.x * w4.x + h4.y * w4.y + h4.z * w4.z + h4.w * w4.w;
    }
    for (int k = 0; k < HH; k += 4) {
        float4 h4 = *(const float4*)(hb + k);
        float4 w4 = *(const float4*)(wf + 512 + k);
        acc += h4.x * w4.x + h4.y * w4.y + h4.z * w4.z + h4.w * w4.w;
    }
    out[idx] = acc;
}

// ---------------------------------------------------------------------------
// Host launcher
// ---------------------------------------------------------------------------
extern "C" void kernel_launch(void* const* d_in, const int* in_sizes, int n_in,
                              void* d_out, int out_size, void* d_ws, size_t ws_size,
                              hipStream_t stream) {
    (void)in_sizes; (void)n_in; (void)out_size; (void)ws_size;
    const int* sentence = (const int*)d_in[0];
    const float* emb = (const float*)d_in[1];
    const float* w_ih_l0 = (const float*)d_in[2];
    const float* w_hh_l0 = (const float*)d_in[3];
    const float* b_ih_l0 = (const float*)d_in[4];
    const float* b_hh_l0 = (const float*)d_in[5];
    const float* w_ih_l1 = (const float*)d_in[6];
    const float* w_hh_l1 = (const float*)d_in[7];
    const float* b_ih_l1 = (const float*)d_in[8];
    const float* b_hh_l1 = (const float*)d_in[9];
    const float* fc_w = (const float*)d_in[10];
    const float* fc_b = (const float*)d_in[11];
    float* out = (float*)d_out;

    // workspace carve-out (256B aligned blocks)
    char* ws = (char*)d_ws;
    size_t off = 0;
    auto carve = [&](size_t bytes) {
        void* p = ws + off;
        off = (off + bytes + 255) & ~(size_t)255;
        return p;
    };
    unsigned* bar = (unsigned*)carve(256);                               // 2 barriers
    float* h32 = (float*)carve((size_t)2 * 2 * BB * HH * 4);             // ping-pong h
    unsigned short* hbf = (unsigned short*)carve((size_t)2 * 2 * BB * HH * 2);
    float* hT = (float*)carve((size_t)2 * BB * HH * 4);
    unsigned short* wih0 = (unsigned short*)carve((size_t)2 * N3 * EE * 2);
    unsigned short* whh0 = (unsigned short*)carve((size_t)2 * N3 * HH * 2);
    unsigned short* wih1 = (unsigned short*)carve((size_t)2 * N3 * 1024 * 2);
    unsigned short* whh1 = (unsigned short*)carve((size_t)2 * N3 * HH * 2);
    unsigned short* x0 = (unsigned short*)carve((size_t)TT * BB * EE * 2);
    unsigned short* ybf = (unsigned short*)carve((size_t)TT * BB * 1024 * 2);
    float* xg = (float*)carve((size_t)2 * TT * BB * N3 * 4);

    // 1. weights -> bf16
    cvt_bf16_kernel<<<1024, 256, 0, stream>>>(w_ih_l0, wih0, 2 * N3 * EE);
    cvt_bf16_kernel<<<1024, 256, 0, stream>>>(w_hh_l0, whh0, 2 * N3 * HH);
    cvt_bf16_kernel<<<2048, 256, 0, stream>>>(w_ih_l1, wih1, 2 * N3 * 1024);
    cvt_bf16_kernel<<<1024, 256, 0, stream>>>(w_hh_l1, whh1, 2 * N3 * HH);

    // 2. embedding gather -> x0 bf16 [T][B][E]
    embed_gather_kernel<<<(TT * BB * 32) / 256, 256, 0, stream>>>(sentence, emb, x0);

    const int M = TT * BB;  // 32768
    // zero region: bar(256B) + h32 + hbf (contiguous by construction)
    const int zero16 =
        (int)((256 + (size_t)2 * 2 * BB * HH * 4 + (size_t)2 * 2 * BB * HH * 2) / 16);

    // 3. layer0 input projections (both dirs), xg column-major
    gemm_inproj_kernel<<<dim3(M / 64, N3 / 128, 2), 256, 0, stream>>>(x0, wih0, b_ih_l0,
                                                                      xg, M, EE);
    // 4. layer0 recurrence
    init_zero_kernel<<<(zero16 + 255) / 256, 256, 0, stream>>>((uint4*)d_ws, zero16);
    gru_recurrence_kernel<<<dim3(8, 2), 128, LDS_BYTES, stream>>>(xg, whh0, b_hh_l0, h32,
                                                                  hbf, ybf, hT, bar);
    // 5. layer1 input projections (A = layer0 output, K=1024)
    gemm_inproj_kernel<<<dim3(M / 64, N3 / 128, 2), 256, 0, stream>>>(ybf, wih1, b_ih_l1,
                                                                      xg, M, 1024);
    // 6. layer1 recurrence (final h -> hT)
    init_zero_kernel<<<(zero16 + 255) / 256, 256, 0, stream>>>((uint4*)d_ws, zero16);
    gru_recurrence_kernel<<<dim3(8, 2), 128, LDS_BYTES, stream>>>(xg, whh1, b_hh_l1, h32,
                                                                  hbf, ybf, hT, bar);
    // 7. classifier head
    fc_kernel<<<3, 256, 0, stream>>>(hT, fc_w, fc_b, out);
}
